// _SpatialGATEncoder_17282948399826
// MI455X (gfx1250) — matrix operand order repaired and checked by reading, not compiled, not run on hardware
//
#include <hip/hip_runtime.h>
#include <hip/hip_bf16.h>

typedef __attribute__((ext_vector_type(16))) _Float16 v16h;
typedef __attribute__((ext_vector_type(8)))  _Float16 v8h;
typedef __attribute__((ext_vector_type(4)))  _Float16 v4h;
typedef __attribute__((ext_vector_type(8)))  float    v8f;

#define HIDDEN    128
#define HEADS     4
#define HEAD_DIM  32
#define NCELLS    50000
#define LN_EPS    1e-6f
#define NEG_SLOPE 0.2f

union AFrag { v16h v; v8h h[2]; };
union CFrag { v8f  v; float e[8]; };
union H8    { v8h  v; _Float16 e[8]; };
union H4    { v4h  v; _Float16 e[4]; };

// ---------------------------------------------------------------------------
// WMMA GEMM: C[M x 128] = A[M x K] * W[K x 128] (+ bias), f32 in/out,
// f16 multiply / f32 accumulate via v_wmma_f32_16x16x32_f16.
// 128 threads = 4 waves; wave w computes rows [blk*64 + 16w, +16), all 128 cols.
// Double-buffered LDS: global loads for chunk c+1 are issued before the WMMA
// burst on chunk c, so VMEM latency hides behind matrix ops.
// Requires K % 4 == 0 (true for 2000 and 128).
// ---------------------------------------------------------------------------
__global__ __launch_bounds__(128) void wmma_gemm128(
    const float* __restrict__ A, const float* __restrict__ W,
    const float* __restrict__ bias, float* __restrict__ C, int M, int K)
{
  __shared__ __align__(16) _Float16 Asub[2][64 * 32];   // [buf][row][k]
  __shared__ __align__(16) _Float16 Wsub[2][128 * 32];  // [buf][col][k]

  const int t    = threadIdx.x;
  const int lane = t & 31;
  const int wv   = t >> 5;
  const int g    = lane >> 4;   // half-wave group
  const int mn   = lane & 15;   // row (A) / col (B,C) within tile
  const int rowBlk = blockIdx.x * 64;
  const int nChunks = (K + 31) / 32;

  CFrag acc[8];
#pragma unroll
  for (int i = 0; i < 8; ++i) { v8f z = {}; acc[i].v = z; }

  // staged-load registers
  float4 aReg[4];
  float  wReg[32];

  // issue all global loads for chunk at k0 (no waits here)
  auto issue_loads = [&](int k0) {
#pragma unroll
    for (int i = 0; i < 4; ++i) {
      const int idx = t + 128 * i;          // 0..511
      const int r   = idx >> 3;             // 0..63
      const int kq  = idx & 7;              // float4 group
      const int R   = rowBlk + r;
      const int Rc  = (R < M) ? R : (M - 1);            // clamp (OOB rows unused)
      const int kidx = k0 + 4 * kq;
      const int kc  = (kidx + 4 <= K) ? kidx : (K - 4); // clamp address
      aReg[i] = *(const float4*)(A + (long)Rc * K + kc);
    }
#pragma unroll
    for (int j = 0; j < 32; ++j) {
      const int kidx = k0 + j;
      const int kc   = (kidx < K) ? kidx : (K - 1);
      wReg[j] = W[(long)kc * HIDDEN + t];
    }
  };

  // convert + select + store staged registers into LDS buffer `buf`
  auto store_tile = [&](int k0, int buf) {
    _Float16* As = &Asub[buf][0];
    _Float16* Ws = &Wsub[buf][0];
#pragma unroll
    for (int i = 0; i < 4; ++i) {
      const int idx = t + 128 * i;
      const int r   = idx >> 3;
      const int kq  = idx & 7;
      const bool kv = (k0 + 4 * kq + 4 <= K);
      H4 hv;
      hv.e[0] = (_Float16)(kv ? aReg[i].x : 0.f);
      hv.e[1] = (_Float16)(kv ? aReg[i].y : 0.f);
      hv.e[2] = (_Float16)(kv ? aReg[i].z : 0.f);
      hv.e[3] = (_Float16)(kv ? aReg[i].w : 0.f);
      *(v4h*)&As[r * 32 + 4 * kq] = hv.v;               // ds_store_b64
    }
#pragma unroll
    for (int grp = 0; grp < 4; ++grp) {
      H8 hv;
#pragma unroll
      for (int j = 0; j < 8; ++j) {
        const bool kv = (k0 + grp * 8 + j < K);
        hv.e[j] = (_Float16)(kv ? wReg[grp * 8 + j] : 0.f);
      }
      *(v8h*)&Ws[t * 32 + grp * 8] = hv.v;              // ds_store_b128
    }
  };

  // compute the 8-WMMA burst from LDS buffer `buf`
  auto compute = [&](int buf) {
    // A fragment: lane(g, m=mn) element e <-> K = 16*(e>>3) + 8*g + (e&7)
    AFrag a;
    const _Float16* arow = &Asub[buf][(wv * 16 + mn) * 32];
    a.h[0] = *(const v8h*)(arow + 8 * g);
    a.h[1] = *(const v8h*)(arow + 16 + 8 * g);
#pragma unroll
    for (int nt = 0; nt < 8; ++nt) {
      // B fragment: lane(g, n=mn) element e <-> K = 16*(e>>3) + 8*g + (e&7) ([col][k])
      AFrag b;
      const _Float16* bcol = &Wsub[buf][(nt * 16 + mn) * 32];
      b.h[0] = *(const v8h*)(bcol + 8 * g);
      b.h[1] = *(const v8h*)(bcol + 16 + 8 * g);
      acc[nt].v = __builtin_amdgcn_wmma_f32_16x16x32_f16(
          false, a.v, false, b.v, (short)0, acc[nt].v, false, false);
    }
  };

  // prologue: stage chunk 0 into buffer 0
  issue_loads(0);
  store_tile(0, 0);
  __syncthreads();

  for (int c = 0; c < nChunks; ++c) {
    const int cur = c & 1;
    const bool more = (c + 1 < nChunks);
    if (more) issue_loads((c + 1) * 32);   // loads in flight during compute
    compute(cur);                          // WMMA burst (dscnt waits only)
    if (more) store_tile((c + 1) * 32, cur ^ 1);  // waits loads, fills next buf
    __syncthreads();
  }

  // ---- epilogue: C/D layout -> row = v + 8g, col = mn ----
#pragma unroll
  for (int nt = 0; nt < 8; ++nt) {
    const int col = nt * 16 + mn;
    const float bv = bias ? bias[col] : 0.f;
#pragma unroll
    for (int v = 0; v < 8; ++v) {
      const int row = rowBlk + wv * 16 + v + 8 * g;
      if (row < M) C[(long)row * HIDDEN + col] = acc[nt].e[v] + bv;
    }
  }
}

// ---------------------------------------------------------------------------
// helpers
// ---------------------------------------------------------------------------
__device__ inline float atomicMaxF(float* addr, float val) {
  if (val >= 0.f)
    return __int_as_float(atomicMax((int*)addr, __float_as_int(val)));
  return __uint_as_float(atomicMin((unsigned int*)addr, __float_as_uint(val)));
}

__global__ void fill_f32(float* __restrict__ p, int n, float v) {
  int i = blockIdx.x * blockDim.x + threadIdx.x;
  if (i < n) p[i] = v;
}

// ---------------------------------------------------------------------------
// edge pass 1: logits[e,h] = attn_h . leaky_relu(hs[src]+hd[dst]+w_e*wedge)
//              and segment-max over dst  (1 thread = (edge, head))
// ---------------------------------------------------------------------------
__global__ void gat_logits(const float* __restrict__ hs, const float* __restrict__ hd,
                           const int* __restrict__ ei, const float* __restrict__ ew,
                           const float* __restrict__ wedge, const float* __restrict__ attn,
                           float* __restrict__ logits, float* __restrict__ mbuf, int E)
{
  int t = blockIdx.x * blockDim.x + threadIdx.x;
  int e = t >> 2, h = t & 3;
  if (e >= E) return;
  int src = ei[e], dst = ei[E + e];
  float we = ew[e];
  const float* ps = hs + (long)src * HIDDEN + h * HEAD_DIM;
  const float* pd = hd + (long)dst * HIDDEN + h * HEAD_DIM;
  const float* pw = wedge + h * HEAD_DIM;
  const float* pa = attn + h * HEAD_DIM;
  float acc = 0.f;
#pragma unroll
  for (int d = 0; d < HEAD_DIM; ++d) {
    float z = ps[d] + pd[d] + we * pw[d];
    z = (z > 0.f) ? z : NEG_SLOPE * z;
    acc += z * pa[d];
  }
  logits[e * HEADS + h] = acc;
  atomicMaxF(&mbuf[dst * HEADS + h], acc);
}

// edge pass 2: ex = exp(logit - m[dst]); segment-sum into denom
__global__ void gat_expsum(const int* __restrict__ ei, float* __restrict__ logits,
                           const float* __restrict__ mbuf, float* __restrict__ denom, int E)
{
  int t = blockIdx.x * blockDim.x + threadIdx.x;
  int e = t >> 2, h = t & 3;
  if (e >= E) return;
  int dst = ei[E + e];
  float ex = __expf(logits[e * HEADS + h] - mbuf[dst * HEADS + h]);
  logits[e * HEADS + h] = ex;
  atomicAdd(&denom[dst * HEADS + h], ex);
}

// edge pass 3: acc[dst] += scale * (ex/denom[dst]) * hs[src]
__global__ void gat_aggregate(const float* __restrict__ hs, const int* __restrict__ ei,
                              const float* __restrict__ logits, const float* __restrict__ denom,
                              float* __restrict__ acc, const float* __restrict__ alpha_p,
                              int use_alpha, int E)
{
  int t = blockIdx.x * blockDim.x + threadIdx.x;
  int e = t >> 2, h = t & 3;
  if (e >= E) return;
  int src = ei[e], dst = ei[E + e];
  float a = alpha_p[0];
  float scale = use_alpha ? a : (1.f - a);
  float att = logits[e * HEADS + h] / (denom[dst * HEADS + h] + 1e-16f);
  float w = scale * att;
  const float* ps = hs + (long)src * HIDDEN + h * HEAD_DIM;
  float* pd = acc + (long)dst * HIDDEN + h * HEAD_DIM;
#pragma unroll
  for (int d = 0; d < HEAD_DIM; ++d) atomicAdd(&pd[d], w * ps[d]);
}

// finalize: blended biases + ELU + LayerNorm, one 128-thread block per node
__global__ __launch_bounds__(128) void gat_finalize(
    const float* __restrict__ acc, const float* __restrict__ alpha_p,
    const float* __restrict__ gfb, const float* __restrict__ gpb,
    const float* __restrict__ lns, const float* __restrict__ lnb,
    float* __restrict__ out)
{
  __shared__ float red[128];
  const int node = blockIdx.x, c = threadIdx.x;
  const float a = alpha_p[0];
  float x = acc[(long)node * HIDDEN + c] + (1.f - a) * gfb[c] + a * gpb[c];
  x = (x > 0.f) ? x : (__expf(x) - 1.f);  // ELU
  red[c] = x; __syncthreads();
  for (int s = 64; s > 0; s >>= 1) { if (c < s) red[c] += red[c + s]; __syncthreads(); }
  const float mu = red[0] * (1.f / 128.f);
  __syncthreads();
  const float dv = x - mu;
  red[c] = dv * dv; __syncthreads();
  for (int s = 64; s > 0; s >>= 1) { if (c < s) red[c] += red[c + s]; __syncthreads(); }
  const float var = red[0] * (1.f / 128.f);
  out[(long)node * HIDDEN + c] = dv * rsqrtf(var + LN_EPS) * lns[c] + lnb[c];
}

// ---------------------------------------------------------------------------
extern "C" void kernel_launch(void* const* d_in, const int* in_sizes, int n_in,
                              void* d_out, int out_size, void* d_ws, size_t ws_size,
                              hipStream_t stream) {
  const float* nf       = (const float*)d_in[0];
  const int*   eif      = (const int*)d_in[1];
  const float* ewf      = (const float*)d_in[2];
  const int*   eip      = (const int*)d_in[3];
  const float* ewp      = (const float*)d_in[4];
  const float* alpha    = (const float*)d_in[5];
  const float* ip_w     = (const float*)d_in[6];
  const float* ip_b     = (const float*)d_in[7];
  const float* gf_wsrc  = (const float*)d_in[8];
  const float* gf_wdst  = (const float*)d_in[9];
  const float* gf_wedge = (const float*)d_in[10];
  const float* gf_attn  = (const float*)d_in[11];
  const float* gf_bias  = (const float*)d_in[12];
  const float* gp_wsrc  = (const float*)d_in[13];
  const float* gp_wdst  = (const float*)d_in[14];
  const float* gp_wedge = (const float*)d_in[15];
  const float* gp_attn  = (const float*)d_in[16];
  const float* gp_bias  = (const float*)d_in[17];
  const float* ln_s     = (const float*)d_in[18];
  const float* ln_b     = (const float*)d_in[19];
  float* out = (float*)d_out;

  // workspace layout (floats); hs/hd/logits reused across the two graphs
  float* ws     = (float*)d_ws;
  float* h      = ws;             // 6,400,000
  float* hs     = ws + 6400000;   // 6,400,000
  float* hd     = ws + 12800000;  // 6,400,000
  float* accb   = ws + 19200000;  // 6,400,000
  float* logits = ws + 25600000;  // 3,200,000 (covers E_F*4)
  float* mbuf   = ws + 28800000;  // 200,000
  float* denom  = ws + 29000000;  // 200,000

  const int E_F = 800000, E_P = 400000;
  const int gemmGrid = (NCELLS + 63) / 64;

  // h = node_features @ ip_w + ip_b
  wmma_gemm128<<<gemmGrid, 128, 0, stream>>>(nf, ip_w, ip_b, h, NCELLS, 2000);
  fill_f32<<<(NCELLS * HIDDEN + 255) / 256, 256, 0, stream>>>(accb, NCELLS * HIDDEN, 0.f);

  // ---- full graph (weight 1 - alpha) ----
  wmma_gemm128<<<gemmGrid, 128, 0, stream>>>(h, gf_wsrc, nullptr, hs, NCELLS, HIDDEN);
  wmma_gemm128<<<gemmGrid, 128, 0, stream>>>(h, gf_wdst, nullptr, hd, NCELLS, HIDDEN);
  fill_f32<<<(NCELLS * HEADS + 255) / 256, 256, 0, stream>>>(mbuf, NCELLS * HEADS, -1e30f);
  fill_f32<<<(NCELLS * HEADS + 255) / 256, 256, 0, stream>>>(denom, NCELLS * HEADS, 0.f);
  gat_logits<<<(E_F * HEADS + 255) / 256, 256, 0, stream>>>(hs, hd, eif, ewf, gf_wedge, gf_attn, logits, mbuf, E_F);
  gat_expsum<<<(E_F * HEADS + 255) / 256, 256, 0, stream>>>(eif, logits, mbuf, denom, E_F);
  gat_aggregate<<<(E_F * HEADS + 255) / 256, 256, 0, stream>>>(hs, eif, logits, denom, accb, alpha, 0, E_F);

  // ---- pruned graph (weight alpha), reusing hs/hd/logits/m/denom ----
  wmma_gemm128<<<gemmGrid, 128, 0, stream>>>(h, gp_wsrc, nullptr, hs, NCELLS, HIDDEN);
  wmma_gemm128<<<gemmGrid, 128, 0, stream>>>(h, gp_wdst, nullptr, hd, NCELLS, HIDDEN);
  fill_f32<<<(NCELLS * HEADS + 255) / 256, 256, 0, stream>>>(mbuf, NCELLS * HEADS, -1e30f);
  fill_f32<<<(NCELLS * HEADS + 255) / 256, 256, 0, stream>>>(denom, NCELLS * HEADS, 0.f);
  gat_logits<<<(E_P * HEADS + 255) / 256, 256, 0, stream>>>(hs, hd, eip, ewp, gp_wedge, gp_attn, logits, mbuf, E_P);
  gat_expsum<<<(E_P * HEADS + 255) / 256, 256, 0, stream>>>(eip, logits, mbuf, denom, E_P);
  gat_aggregate<<<(E_P * HEADS + 255) / 256, 256, 0, stream>>>(hs, eip, logits, denom, accb, alpha, 1, E_P);

  // ELU + LayerNorm
  gat_finalize<<<NCELLS, 128, 0, stream>>>(accb, alpha, gf_bias, gp_bias, ln_s, ln_b, out);
}
